// Encoder_Block_44538810859673
// MI455X (gfx1250) — matrix operand order repaired and checked by reading, not compiled, not run on hardware
//
#include <hip/hip_runtime.h>
#include <hip/hip_bf16.h>

#define DIMC 256
#define NNODE 128
#define HID 1024
#define LNEPS 1e-5f
#define USE_TDM 1

typedef __attribute__((ext_vector_type(16))) _Float16 v16h;
typedef __attribute__((ext_vector_type(8)))  float    v8f;
typedef _Float16 h16;

typedef unsigned int u32x4 __attribute__((ext_vector_type(4)));
typedef int          i32x8 __attribute__((ext_vector_type(8)));
typedef int          i32x4 __attribute__((ext_vector_type(4)));

// ---------------- TDM: 2D f32 tile (rows x cols, row stride) global -> LDS ----
__device__ __forceinline__ void tdm_load_2d_f32(const float* gptr, unsigned lds_off,
                                                int rows, int cols, int stride) {
#if USE_TDM
  unsigned long long ga = (unsigned long long)(uintptr_t)gptr;
  u32x4 g0;
  g0[0] = 1u;                                   // count=1, user mode
  g0[1] = lds_off;                              // lds_addr (bytes)
  g0[2] = (unsigned)(ga & 0xFFFFFFFFu);         // global_addr[31:0]
  g0[3] = (unsigned)((ga >> 32) & 0x01FFFFFFu) | (2u << 30);  // ga[56:32] | type=2
  i32x8 g1;
  g1[0] = (int)(2u << 16);                      // data_size=2 (4 bytes)
  g1[1] = (int)((unsigned)(cols & 0xFFFF) << 16);             // tensor_dim0 lo16
  g1[2] = (int)(((unsigned)cols >> 16) | ((unsigned)(rows & 0xFFFF) << 16)); // dim0 hi | dim1 lo
  g1[3] = (int)(((unsigned)rows >> 16) | ((unsigned)(cols & 0xFFFF) << 16)); // dim1 hi | tile_dim0
  g1[4] = (int)((unsigned)(rows & 0xFFFF));     // tile_dim1 ; tile_dim2=0
  g1[5] = (int)(unsigned)stride;                // tensor_dim0_stride lo32
  g1[6] = 0;
  g1[7] = 0;
  i32x4 z4 = {0, 0, 0, 0};
  i32x8 z8 = {0, 0, 0, 0, 0, 0, 0, 0};
  __builtin_amdgcn_tensor_load_to_lds(g0, g1, z4, z4, z8, 0);
  __builtin_amdgcn_s_wait_tensorcnt(0);
#else
  (void)gptr; (void)lds_off; (void)rows; (void)cols; (void)stride;
#endif
}

// ---------------- WMMA fragment loaders (CDNA5 16x16x32 f16 layouts) ----------
__device__ __forceinline__ v16h frag_a_h16(const h16* base, int ld, int kOff) {
  int lane = threadIdx.x & 31;
  int m = lane & 15;
  int k0 = kOff + ((lane >> 4) << 3);
  const h16* p = base + m * ld + k0;
  v16h a;
#pragma unroll
  for (int i = 0; i < 8; ++i) a[i] = p[i];
#pragma unroll
  for (int i = 0; i < 8; ++i) a[8 + i] = p[16 + i];
  return a;
}

__device__ __forceinline__ v16h frag_a_f32(const float* base, int ld, int kOff) {
  int lane = threadIdx.x & 31;
  int m = lane & 15;
  int k0 = kOff + ((lane >> 4) << 3);
  const float* p = base + m * ld + k0;
  v16h a;
#pragma unroll
  for (int i = 0; i < 8; ++i) a[i] = (h16)p[i];
#pragma unroll
  for (int i = 0; i < 8; ++i) a[8 + i] = (h16)p[16 + i];
  return a;
}

__device__ __forceinline__ v16h frag_b_lds(const h16* Wt, int n0, int ld, int kOff) {
  int lane = threadIdx.x & 31;
  const h16* p = Wt + (n0 + (lane & 15)) * ld + kOff + ((lane >> 4) << 3);
  v16h b;
#pragma unroll
  for (int i = 0; i < 8; ++i) b[i] = p[i];
#pragma unroll
  for (int i = 0; i < 8; ++i) b[8 + i] = p[16 + i];
  return b;
}

__device__ __forceinline__ float wave_sum(float v) {
#pragma unroll
  for (int off = 16; off > 0; off >>= 1) v += __shfl_xor(v, off, 32);
  return v;
}

__device__ __forceinline__ float block_sum(float v, float* scratch) { // blockDim==256
  v = wave_sum(v);
  __syncthreads();
  if ((threadIdx.x & 31) == 0) scratch[threadIdx.x >> 5] = v;
  __syncthreads();
  float s = 0.f;
#pragma unroll
  for (int q = 0; q < 8; ++q) s += scratch[q];
  return s;
}

// ---------------- K0: cast weights to f16, transposed to [N][K] ----------------
__global__ void k_cast_transpose(const float* __restrict__ in, h16* __restrict__ out,
                                 int K, int Nn) {
  int idx = blockIdx.x * blockDim.x + threadIdx.x;
  if (idx >= K * Nn) return;
  int nn = idx / K, kk = idx - nn * K;
  out[idx] = (h16)in[(size_t)kk * Nn + nn];
}

// ---------------- K1: x1 = LN1(x); q,k,v = x1@W + b  (8 rows / WG) ------------
__global__ void k_x_pre(const float* __restrict__ x,
                        const float* __restrict__ Wq, const float* __restrict__ bq,
                        const float* __restrict__ Wk, const float* __restrict__ bk,
                        const float* __restrict__ Wv, const float* __restrict__ bv,
                        const float* __restrict__ ln1s, const float* __restrict__ ln1b,
                        float* __restrict__ x1, float* __restrict__ qo,
                        float* __restrict__ ko, float* __restrict__ vo) {
  __shared__ float sX1[8 * DIMC];
  int row0 = blockIdx.x * 8;
  int tid = threadIdx.x, w = tid >> 5, lane = tid & 31;
  {  // wave w LayerNorms row row0+w
    const float* xrow = x + (size_t)(row0 + w) * DIMC;
    int c0 = lane * 8;
    float xv[8]; float s = 0.f, s2 = 0.f;
#pragma unroll
    for (int u = 0; u < 8; ++u) { xv[u] = xrow[c0 + u]; s += xv[u]; s2 += xv[u] * xv[u]; }
    s = wave_sum(s); s2 = wave_sum(s2);
    float m = s * (1.f / DIMC), var = s2 * (1.f / DIMC) - m * m;
    float rs = rsqrtf(var + LNEPS);
#pragma unroll
    for (int u = 0; u < 8; ++u) {
      float t = (xv[u] - m) * rs * ln1s[c0 + u] + ln1b[c0 + u];
      sX1[w * DIMC + c0 + u] = t;
      x1[(size_t)(row0 + w) * DIMC + c0 + u] = t;
    }
  }
  __syncthreads();
  int c = tid;
  float aq[8], ak[8], av[8];
#pragma unroll
  for (int r = 0; r < 8; ++r) { aq[r] = bq[c]; ak[r] = bk[c]; av[r] = bv[c]; }
  for (int kk = 0; kk < DIMC; ++kk) {
    float wq = Wq[(size_t)kk * DIMC + c];
    float wk = Wk[(size_t)kk * DIMC + c];
    float wv = Wv[(size_t)kk * DIMC + c];
#pragma unroll
    for (int r = 0; r < 8; ++r) {
      float t = sX1[r * DIMC + kk];
      aq[r] += t * wq; ak[r] += t * wk; av[r] += t * wv;
    }
  }
#pragma unroll
  for (int r = 0; r < 8; ++r) {
    size_t o = (size_t)(row0 + r) * DIMC + c;
    qo[o] = aq[r]; ko[o] = ak[r]; vo[o] = av[r];
  }
}

// ---------------- K2: fused e-GEMM + attn + edge-GEMM + LN4 + softmax/agg ------
__global__ void __launch_bounds__(256, 1)
k_edge_attn(const float* __restrict__ y,
            const float* __restrict__ qg, const float* __restrict__ kg,
            const float* __restrict__ vg,
            const h16* __restrict__ WeT, const float* __restrict__ be,
            const h16* __restrict__ WoeT, const float* __restrict__ boe,
            const float* __restrict__ ln4s, const float* __restrict__ ln4b,
            float* __restrict__ y2out, float* __restrict__ aggout) {
  extern __shared__ char smem[];
  h16*   sAttn = (h16*)smem;                           // 64KB
  float* sY    = (float*)(smem + 65536);               // 128KB (phase1, TDM dest)
  float* sEdge = (float*)(smem + 65536);               // 128KB (overlaps sY)
  h16*   sK    = (h16*)(smem + 65536 + 131072);        // 64KB
  float* sQ    = (float*)(smem + 65536 + 131072 + 65536);       // 1KB
  h16*   sWB   = (h16*)(smem + 65536 + 131072 + 65536 + 1024);  // 8KB staged B tile

  int bi = blockIdx.x;            // b*128 + i
  int b  = bi >> 7;
  int tid = threadIdx.x;
  size_t ybase = (size_t)bi * NNODE * DIMC;

  if (tid < 32)
    tdm_load_2d_f32(y + ybase, (unsigned)(uintptr_t)sY, NNODE, DIMC, DIMC);
#if !USE_TDM
  for (int idx = tid; idx < NNODE * DIMC; idx += 256) sY[idx] = y[ybase + idx];
#endif
  const float* kbase = kg + (size_t)b * NNODE * DIMC;
  for (int idx = tid; idx < NNODE * DIMC; idx += 256) sK[idx] = (h16)kbase[idx];
  sQ[tid] = qg[(size_t)bi * DIMC + tid];
  __syncthreads();

  int w = tid >> 5, lane = tid & 31;
  int jb = w * 16;
  int n = lane & 15, hi = lane >> 4;

  // ---- phase 1: e = y@We + be ; attn = q*k/sqrt(32)*(e+1)*e -> sAttn ----
  v16h af[8];
#pragma unroll
  for (int kt = 0; kt < 8; ++kt) af[kt] = frag_a_f32(sY + jb * DIMC, DIMC, kt * 32);

  for (int nt = 0; nt < 16; ++nt) {
    __syncthreads();
    { int n0 = tid >> 4, kk0 = (tid & 15) << 4;
      const uint4* s = (const uint4*)(WeT + ((size_t)(nt * 16 + n0) << 8) + kk0);
      uint4* d = (uint4*)(sWB + (n0 << 8) + kk0);
      d[0] = s[0]; d[1] = s[1]; }
    __syncthreads();
    v8f acc = {};
#pragma unroll
    for (int kt = 0; kt < 8; ++kt)
      acc = __builtin_amdgcn_wmma_f32_16x16x32_f16(false, af[kt], false,
              frag_b_lds(sWB, 0, DIMC, kt * 32), (short)0, acc, false, false);
    int c = nt * 16 + n;
    float bec = be[c], qv = sQ[c];
#pragma unroll
    for (int r = 0; r < 8; ++r) {
      int j = jb + r + 8 * hi;
      float e = acc[r] + bec;
      float kv = (float)sK[j * DIMC + c];
      float att = qv * kv * 0.17677669529663687f * (e + 1.f) * e;
      sAttn[j * DIMC + c] = (h16)att;
    }
  }
  __syncthreads();

  // ---- phase 2: edge_out = attn@Woe + boe -> sEdge ----
#pragma unroll
  for (int kt = 0; kt < 8; ++kt) af[kt] = frag_a_h16(sAttn + jb * DIMC, DIMC, kt * 32);

  for (int nt = 0; nt < 16; ++nt) {
    __syncthreads();
    { int n0 = tid >> 4, kk0 = (tid & 15) << 4;
      const uint4* s = (const uint4*)(WoeT + ((size_t)(nt * 16 + n0) << 8) + kk0);
      uint4* d = (uint4*)(sWB + (n0 << 8) + kk0);
      d[0] = s[0]; d[1] = s[1]; }
    __syncthreads();
    v8f acc = {};
#pragma unroll
    for (int kt = 0; kt < 8; ++kt)
      acc = __builtin_amdgcn_wmma_f32_16x16x32_f16(false, af[kt], false,
              frag_b_lds(sWB, 0, DIMC, kt * 32), (short)0, acc, false, false);
    int c = nt * 16 + n;
    float bo = boe[c];
#pragma unroll
    for (int r = 0; r < 8; ++r) {
      int j = jb + r + 8 * hi;
      sEdge[j * DIMC + c] = acc[r] + bo;
    }
  }
  __syncthreads();

  // ---- phase 3: y2 = LN4(edge_out + y) ----
  for (int j = w; j < NNODE; j += 8) {
    const float* yrow = y + ybase + (size_t)j * DIMC;
    float* orow = y2out + ybase + (size_t)j * DIMC;
    int c0 = lane * 8;
    float tv[8]; float s = 0.f, s2 = 0.f;
#pragma unroll
    for (int u = 0; u < 8; ++u) {
      float t = sEdge[j * DIMC + c0 + u] + yrow[c0 + u];
      tv[u] = t; s += t; s2 += t * t;
    }
    s = wave_sum(s); s2 = wave_sum(s2);
    float m = s * (1.f / DIMC), var = s2 * (1.f / DIMC) - m * m;
    float rs = rsqrtf(var + LNEPS);
#pragma unroll
    for (int u = 0; u < 8; ++u)
      orow[c0 + u] = (tv[u] - m) * rs * ln4s[c0 + u] + ln4b[c0 + u];
  }

  // ---- phase 4: per-channel softmax over j + v aggregation ----
  {
    int c = tid;
    float mx = -3.0e38f;
    for (int j = 0; j < NNODE; ++j)
      mx = fmaxf(mx, (float)sAttn[j * DIMC + c]);
    const float* vb = vg + (size_t)b * NNODE * DIMC;
    float Z = 0.f, num = 0.f;
    for (int j = 0; j < NNODE; ++j) {
      float e = expf((float)sAttn[j * DIMC + c] - mx);
      Z += e;
      num += e * vb[(size_t)j * DIMC + c];
    }
    aggout[(size_t)bi * DIMC + c] = num / Z;
  }
}

// ---------------- K3: node_out + LN3 -> x2  (8 rows / WG) ----------------------
__global__ void k_node(const float* __restrict__ x1, const float* __restrict__ agg,
                       const float* __restrict__ Won, const float* __restrict__ bon,
                       const float* __restrict__ ln3s, const float* __restrict__ ln3b,
                       float* __restrict__ x2out) {
  __shared__ float sA[8 * DIMC];
  __shared__ float sred[8];
  int row0 = blockIdx.x * 8;
  int tid = threadIdx.x;
  for (int idx = tid; idx < 8 * DIMC; idx += 256) sA[idx] = agg[(size_t)row0 * DIMC + idx];
  __syncthreads();
  int c = tid;
  float acc[8];
#pragma unroll
  for (int r = 0; r < 8; ++r) acc[r] = bon[c];
  for (int kk = 0; kk < DIMC; ++kk) {
    float wv = Won[(size_t)kk * DIMC + c];
#pragma unroll
    for (int r = 0; r < 8; ++r) acc[r] += sA[r * DIMC + kk] * wv;
  }
#pragma unroll
  for (int r = 0; r < 8; ++r) {
    float t = x1[(size_t)(row0 + r) * DIMC + c] + acc[r];
    float s = block_sum(t, sred), s2 = block_sum(t * t, sred);
    float m = s * (1.f / DIMC), var = s2 * (1.f / DIMC) - m * m;
    x2out[(size_t)(row0 + r) * DIMC + c] =
        (t - m) * rsqrtf(var + LNEPS) * ln3s[c] + ln3b[c];
  }
}

// ---------------- K4: generic fused  dst = LN(src + relu(src@W1+b1)@W2+b2) -----
// grid = (#rows / 128); WMMA with per-ht LDS-staged weights, register A strip.
__global__ void __launch_bounds__(256, 1)
k_mlp_ln(const float* __restrict__ src,
         const h16* __restrict__ W1T, const float* __restrict__ b1,
         const h16* __restrict__ W2T, const float* __restrict__ b2,
         const float* __restrict__ lnS, const float* __restrict__ lnB,
         float* __restrict__ dst) {
  extern __shared__ char smem[];
  float* sX   = (float*)smem;                    // 128KB (start, TDM dest)
  float* sOut = (float*)smem;                    // 128KB (end, overlaps sX)
  h16*   hbuf = (h16*)(smem + 131072);           // 8KB  (per-wave 16x32)
  h16*   sW1t = (h16*)(smem + 131072 + 8192);    // 16KB (32 cols x 256 k)
  h16*   sW2t = (h16*)(smem + 131072 + 8192 + 16384); // 16KB (256 cols x 32 k)

  int bi = blockIdx.x;
  size_t base = (size_t)bi * NNODE * DIMC;
  int tid = threadIdx.x;
  if (tid < 32)
    tdm_load_2d_f32(src + base, (unsigned)(uintptr_t)sX, NNODE, DIMC, DIMC);
#if !USE_TDM
  for (int idx = tid; idx < NNODE * DIMC; idx += 256) sX[idx] = src[base + idx];
#endif
  __syncthreads();

  int w = tid >> 5, lane = tid & 31;
  int jb = w * 16;
  int n = lane & 15, hi = lane >> 4;
  h16* hb = hbuf + w * 512;

  v16h afX[8];
#pragma unroll
  for (int kt = 0; kt < 8; ++kt) afX[kt] = frag_a_f32(sX + jb * DIMC, DIMC, kt * 32);

  v8f outacc[16];
#pragma unroll
  for (int q = 0; q < 16; ++q) { v8f z = {}; outacc[q] = z; }

  for (int ht = 0; ht < 32; ++ht) {
    __syncthreads();   // previous iteration's weight-tile readers done
    { // stage W1 tile: 32 hidden cols x 256 k
      int n0 = tid >> 3, kseg = (tid & 7) << 5;
      const uint4* s = (const uint4*)(W1T + ((size_t)(ht * 32 + n0) << 8) + kseg);
      uint4* d = (uint4*)(sW1t + (n0 << 8) + kseg);
      d[0] = s[0]; d[1] = s[1]; d[2] = s[2]; d[3] = s[3];
    }
    { // stage W2 tile: 256 out cols x 32 k
      const uint4* s = (const uint4*)(W2T + ((size_t)tid << 10) + ht * 32);
      uint4* d = (uint4*)(sW2t + (tid << 5));
      d[0] = s[0]; d[1] = s[1]; d[2] = s[2]; d[3] = s[3];
    }
    __syncthreads();

#pragma unroll
    for (int hnt = 0; hnt < 2; ++hnt) {
      v8f hc = {};
#pragma unroll
      for (int kt = 0; kt < 8; ++kt)
        hc = __builtin_amdgcn_wmma_f32_16x16x32_f16(false, afX[kt], false,
               frag_b_lds(sW1t, hnt * 16, DIMC, kt * 32), (short)0, hc, false, false);
      int hh = ht * 32 + hnt * 16 + n;
      float bb1 = b1[hh];
#pragma unroll
      for (int r = 0; r < 8; ++r) {
        int m = r + 8 * hi;
        hb[m * 32 + hnt * 16 + n] = (h16)fmaxf(hc[r] + bb1, 0.f);
      }
    }
    __asm__ volatile("s_wait_dscnt 0" ::: "memory"); // hb stores visible (in-wave)
    v16h afH = frag_a_h16(hb, 32, 0);
#pragma unroll
    for (int nt = 0; nt < 16; ++nt)
      outacc[nt] = __builtin_amdgcn_wmma_f32_16x16x32_f16(false, afH, false,
                     frag_b_lds(sW2t, nt * 16, 32, 0), (short)0, outacc[nt], false, false);
  }

#pragma unroll
  for (int nt = 0; nt < 16; ++nt) {
    int c = nt * 16 + n;
    float bb2 = b2[c];
#pragma unroll
    for (int r = 0; r < 8; ++r) {
      int j = jb + r + 8 * hi;
      sOut[j * DIMC + c] = outacc[nt][r] + bb2;
    }
  }
  __syncthreads();

  for (int j = w; j < NNODE; j += 8) {
    const float* srow = src + base + (size_t)j * DIMC;
    float* orow = dst + base + (size_t)j * DIMC;
    int c0 = lane * 8;
    float tv[8]; float s = 0.f, s2 = 0.f;
#pragma unroll
    for (int u = 0; u < 8; ++u) {
      float t = sOut[j * DIMC + c0 + u] + srow[c0 + u];
      tv[u] = t; s += t; s2 += t * t;
    }
    s = wave_sum(s); s2 = wave_sum(s2);
    float m = s * (1.f / DIMC), var = s2 * (1.f / DIMC) - m * m;
    float rs = rsqrtf(var + LNEPS);
#pragma unroll
    for (int u = 0; u < 8; ++u)
      orow[c0 + u] = (tv[u] - m) * rs * lnS[c0 + u] + lnB[c0 + u];
  }
}

// ---------------- launcher ----------------
extern "C" void kernel_launch(void* const* d_in, const int* in_sizes, int n_in,
                              void* d_out, int out_size, void* d_ws, size_t ws_size,
                              hipStream_t stream) {
  (void)in_sizes; (void)n_in; (void)out_size; (void)ws_size;
  const float* x    = (const float*)d_in[0];
  const float* y    = (const float*)d_in[1];
  const float* Wq   = (const float*)d_in[2];  const float* bq  = (const float*)d_in[3];
  const float* Wk   = (const float*)d_in[4];  const float* bk  = (const float*)d_in[5];
  const float* Wv   = (const float*)d_in[6];  const float* bv  = (const float*)d_in[7];
  const float* We   = (const float*)d_in[8];  const float* be  = (const float*)d_in[9];
  const float* Woe  = (const float*)d_in[10]; const float* boe = (const float*)d_in[11];
  const float* Won  = (const float*)d_in[12]; const float* bon = (const float*)d_in[13];
  const float* m1w1 = (const float*)d_in[14]; const float* m1b1 = (const float*)d_in[15];
  const float* m1w2 = (const float*)d_in[16]; const float* m1b2 = (const float*)d_in[17];
  const float* m2w1 = (const float*)d_in[18]; const float* m2b1 = (const float*)d_in[19];
  const float* m2w2 = (const float*)d_in[20]; const float* m2b2 = (const float*)d_in[21];
  const float* ln1s = (const float*)d_in[22]; const float* ln1b = (const float*)d_in[23];
  const float* ln3s = (const float*)d_in[24]; const float* ln3b = (const float*)d_in[25];
  const float* ln4s = (const float*)d_in[26]; const float* ln4b = (const float*)d_in[27];
  const float* ln5s = (const float*)d_in[28]; const float* ln5b = (const float*)d_in[29];
  const float* ln6s = (const float*)d_in[30]; const float* ln6b = (const float*)d_in[31];

  float* xout = (float*)d_out;           // 8*128*256
  float* yreg = (float*)d_out + 262144;  // y2 then y_out (in place)

  char* ws = (char*)d_ws;
  float* x1  = (float*)(ws + 0 * 1048576);
  float* qb  = (float*)(ws + 1 * 1048576);
  float* kb  = (float*)(ws + 2 * 1048576);
  float* vb  = (float*)(ws + 3 * 1048576);
  float* agg = (float*)(ws + 4 * 1048576);
  float* x2b = (float*)(ws + 5 * 1048576);
  h16* WeT  = (h16*)(ws + 6 * 1048576);
  h16* WoeT = (h16*)(ws + 6 * 1048576 + 131072);
  h16* W1xT = (h16*)(ws + 6 * 1048576 + 262144);
  h16* W2xT = (h16*)(ws + 6 * 1048576 + 262144 + 524288);
  h16* W1yT = (h16*)(ws + 6 * 1048576 + 262144 + 1048576);
  h16* W2yT = (h16*)(ws + 6 * 1048576 + 262144 + 1572864);

  k_cast_transpose<<<(256 * 256 + 255) / 256, 256, 0, stream>>>(We,   WeT,  256, 256);
  k_cast_transpose<<<(256 * 256 + 255) / 256, 256, 0, stream>>>(Woe,  WoeT, 256, 256);
  k_cast_transpose<<<(256 * 1024 + 255) / 256, 256, 0, stream>>>(m1w1, W1xT, 256, 1024);
  k_cast_transpose<<<(1024 * 256 + 255) / 256, 256, 0, stream>>>(m1w2, W2xT, 1024, 256);
  k_cast_transpose<<<(256 * 1024 + 255) / 256, 256, 0, stream>>>(m2w1, W1yT, 256, 1024);
  k_cast_transpose<<<(1024 * 256 + 255) / 256, 256, 0, stream>>>(m2w2, W2yT, 1024, 256);

  k_x_pre<<<128, 256, 0, stream>>>(x, Wq, bq, Wk, bk, Wv, bv, ln1s, ln1b,
                                   x1, qb, kb, vb);

  size_t smem2 = 65536 + 131072 + 65536 + 1024 + 8192;  // 271360 (< 320KB/WGP)
  k_edge_attn<<<1024, 256, smem2, stream>>>(y, qb, kb, vb, WeT, be, WoeT, boe,
                                            ln4s, ln4b, yreg, agg);

  k_node<<<128, 256, 0, stream>>>(x1, agg, Won, bon, ln3s, ln3b, x2b);

  size_t smem4 = 131072 + 8192 + 16384 + 16384;         // 172032
  // x path: 1024 rows = 8 slices
  k_mlp_ln<<<8, 256, smem4, stream>>>(x2b, W1xT, m1b1, W2xT, m1b2, ln5s, ln5b, xout);
  // y path: 131072 rows = 1024 slices (in place over d_out y region)
  k_mlp_ln<<<1024, 256, smem4, stream>>>(yreg, W1yT, m2b1, W2yT, m2b2, ln6s, ln6b, yreg);
}
